// BaselineNCDE_64115271794909
// MI455X (gfx1250) — matrix operand order repaired and checked by reading, not compiled, not run on hardware
//
#include <hip/hip_runtime.h>
#include <hip/hip_bf16.h>

// ---------------- problem constants (from reference) ----------------
#define BATCH 4096
#define TT    256
#define FF    6
#define HH    64
#define WW    128
#define CC    7        // F + 1
#define TB    16       // batch rows per workgroup
#define NTHREADS 256   // 8 waves of 32

typedef __attribute__((ext_vector_type(16))) __bf16 v16bf;
typedef __attribute__((ext_vector_type(8)))  float  v8f;

__device__ __forceinline__ unsigned short f32_to_bf16(float f) {
  union { float f; unsigned int u; } v; v.f = f;
  unsigned int u = v.u;
  u += 0x7FFFu + ((u >> 16) & 1u);   // round-to-nearest-even
  return (unsigned short)(u >> 16);
}

// Branch-free tanh: tanh(x) = sign(x) * (1 - 2/(exp(2|x|)+1)).
// exp2 overflow -> inf -> rcp -> 0 -> +/-1, so saturation needs no branch.
__device__ __forceinline__ float fast_tanh(float x) {
  float ax = __builtin_fabsf(x);
  float e  = __builtin_amdgcn_exp2f(ax * 2.885390081777927f);  // 2*log2(e)
  float r  = 1.0f - 2.0f * __builtin_amdgcn_rcpf(e + 1.0f);
  return __builtin_copysignf(r, x);
}

__device__ __forceinline__ float gelu_tanh(float x) {
  // jax.nn.gelu default (approximate=True)
  float x3 = x * x * x;
  return 0.5f * x * (1.0f + fast_tanh(0.7978845608028654f * (x + 0.044715f * x3)));
}

// A-layout scatter index for a 16-row x K bf16 matrix (WMMA 16x16x32 A fragment).
// ISA layout: lane<16 holds row=lane, e<8 -> K=e, e>=8 -> K=e+8 ; lanes>=16: +8 on K.
__device__ __forceinline__ int a_idx(int row, int k) {
  int chunk = k >> 5;
  int kk    = k & 31;
  int lane  = row + (((kk >> 3) & 1) << 4);
  int e     = (kk & 7) | (((kk >> 4) & 1) << 3);
  return ((chunk << 5) + lane) * 16 + e;
}

__device__ __forceinline__ v8f wmma_bf16(const unsigned short* Ap,
                                         const unsigned short* Bp, v8f c) {
  v16bf a = *reinterpret_cast<const v16bf*>(Ap);
  v16bf b = *reinterpret_cast<const v16bf*>(Bp);
  return __builtin_amdgcn_wmma_f32_16x16x32_bf16(false, a, false, b,
                                                 (short)0, c, false, false);
}

// Pre-layout a weight W (N_out x K_in, row-major f32) into LDS as bf16 WMMA
// B-fragments: flat [kc][nt][lane][e], B[k][n] = W[n][k].
__device__ void prelayout_B(const float* __restrict__ Wg, unsigned short* dst,
                            int K, int Nt, int Kc, int tid) {
  int total = Kc * Nt * 512;
  for (int i = tid; i < total; i += NTHREADS) {
    int e = i & 15;
    int lane = (i >> 4) & 31;
    int r = i >> 9;
    int nt = r % Nt;
    int kc = r / Nt;
    int n = nt * 16 + (lane & 15);
    int k = kc * 32 + (e & 7) + ((lane >> 4) << 3) + ((e >> 3) << 4);
    dst[i] = f32_to_bf16(Wg[n * K + k]);
  }
}

__global__ __launch_bounds__(NTHREADS, 1)
void ncde_scan_kernel(const float* __restrict__ timeg,
                      const float* __restrict__ feat,
                      const int*   __restrict__ length,
                      const float* __restrict__ Wp,  const float* __restrict__ bp,
                      const float* __restrict__ W1,  const float* __restrict__ b1,
                      const float* __restrict__ W2,  const float* __restrict__ b2,
                      const float* __restrict__ W3,  const float* __restrict__ b3,
                      const float* __restrict__ Wr1, const float* __restrict__ br1,
                      const float* __restrict__ Wr2, const float* __restrict__ br2,
                      float* __restrict__ out) {
  // -------- LDS: weights resident in WMMA B-layout (bf16), reused all 256 steps
  __shared__ unsigned short W1B[2 * 8 * 32 * 16]  __attribute__((aligned(32))); // 16 KB
  __shared__ unsigned short W2B[4 * 8 * 32 * 16]  __attribute__((aligned(32))); // 32 KB
  __shared__ unsigned short W3B[4 * 28 * 32 * 16] __attribute__((aligned(32))); // 112 KB
  __shared__ unsigned short WrB[2 * 2 * 32 * 16]  __attribute__((aligned(32))); // 4 KB
  // -------- activations
  __shared__ unsigned short hA [2 * 32 * 16] __attribute__((aligned(32)));      // h  (16x64)  A-layout
  __shared__ unsigned short x1A[4 * 32 * 16] __attribute__((aligned(32)));      // x1 (16x128) A-layout
  __shared__ unsigned short x2A[4 * 32 * 16] __attribute__((aligned(32)));      // x2 (16x128) A-layout
  __shared__ float g3[TB * 456];      // tanh(GEMM3) 16x448, padded stride
  __shared__ float h32[TB * 68];      // f32 master copy of h, padded stride
  __shared__ float b1L[WW], b2L[WW], b3L[HH * CC], br1L[32], wr2L[32];
  __shared__ float dxdtL[TB * 8], dtL[TB], actL[TB], predP[2 * TB];
  __shared__ float br2L;

  const int tid   = threadIdx.x;
  const int wv    = tid >> 5;
  const int lane  = tid & 31;
  const int lhalf = lane & 15;
  const int mbase = (lane < 16) ? 0 : 8;       // C-fragment row offset
  const int b0    = blockIdx.x * TB;

  // ---------------- one-time setup ----------------
  for (int i = tid; i < WW; i += NTHREADS) { b1L[i] = b1[i]; b2L[i] = b2[i]; }
  for (int i = tid; i < HH * CC; i += NTHREADS) b3L[i] = b3[i];
  if (tid < 32) { br1L[tid] = br1[tid]; wr2L[tid] = Wr2[tid]; }
  if (tid == 0) br2L = br2[0];

  prelayout_B(W1,  W1B, 64,  8,  2, tid);
  prelayout_B(W2,  W2B, 128, 8,  4, tid);
  prelayout_B(W3,  W3B, 128, 28, 4, tid);
  prelayout_B(Wr1, WrB, 64,  2,  2, tid);

  // h0 = features[:,0] @ Wp.T + bp
  for (int p = tid; p < TB * HH; p += NTHREADS) {
    int row = p >> 6, hh = p & 63;
    const float* fr = feat + (size_t)(b0 + row) * TT * FF;   // t = 0
    float s = bp[hh];
#pragma unroll
    for (int f = 0; f < FF; ++f) s += fr[f] * Wp[hh * FF + f];
    h32[row * 68 + hh] = s;
    hA[a_idx(row, hh)] = f32_to_bf16(s);
  }
  __syncthreads();

  // ---------------- sequential scan over T ----------------
  for (int t = 0; t < TT; ++t) {
    // per-step dX/dt, dt, active  (threads 0..15, one batch row each)
    if (tid < TB) {
      int b  = b0 + tid;
      int tn = (t + 1 < TT) ? t + 1 : TT - 1;
      float tc  = timeg[(size_t)b * TT + t];
      float tnx = timeg[(size_t)b * TT + tn];
      float dt  = fmaxf(tnx - tc, 1e-6f);
      float inv = 1.0f / dt;
      dtL[tid]  = dt;
      actL[tid] = (t < length[b]) ? 1.0f : 0.0f;
      dxdtL[tid * 8 + 0] = (tnx - tc) * inv;              // time channel
      const float* fc = feat + ((size_t)b * TT + t)  * FF;
      const float* fn = feat + ((size_t)b * TT + tn) * FF;
#pragma unroll
      for (int f = 0; f < FF; ++f)
        dxdtL[tid * 8 + 1 + f] = (fn[f] - fc[f]) * inv;
      if (t + 2 < TT)                                      // gfx1250 prefetch
        __builtin_prefetch(feat + ((size_t)b * TT + t + 2) * FF, 0, 1);
    }

    // ---- GEMM1: x1 = gelu(h @ W1.T + b1)   (16x64)x(64x128)
    {
      v8f acc = {};
#pragma unroll
      for (int kc = 0; kc < 2; ++kc)
        acc = wmma_bf16(&hA[(kc * 32 + lane) * 16],
                        &W1B[((kc * 8 + wv) * 32 + lane) * 16], acc);
#pragma unroll
      for (int r = 0; r < 8; ++r) {
        int row = r + mbase, col = wv * 16 + lhalf;
        x1A[a_idx(row, col)] = f32_to_bf16(gelu_tanh(acc[r] + b1L[col]));
      }
    }
    __syncthreads();

    // ---- GEMM2: x2 = gelu(x1 @ W2.T + b2)  (16x128)x(128x128)
    // two independent accumulation chains, merged at the end
    {
      v8f acc0 = {}, acc1 = {};
      acc0 = wmma_bf16(&x1A[(0 * 32 + lane) * 16],
                       &W2B[((0 * 8 + wv) * 32 + lane) * 16], acc0);
      acc1 = wmma_bf16(&x1A[(1 * 32 + lane) * 16],
                       &W2B[((1 * 8 + wv) * 32 + lane) * 16], acc1);
      acc0 = wmma_bf16(&x1A[(2 * 32 + lane) * 16],
                       &W2B[((2 * 8 + wv) * 32 + lane) * 16], acc0);
      acc1 = wmma_bf16(&x1A[(3 * 32 + lane) * 16],
                       &W2B[((3 * 8 + wv) * 32 + lane) * 16], acc1);
#pragma unroll
      for (int r = 0; r < 8; ++r) {
        int row = r + mbase, col = wv * 16 + lhalf;
        float v = gelu_tanh(acc0[r] + acc1[r] + b2L[col]);
        x2A[a_idx(row, col)] = f32_to_bf16(v);
      }
    }
    __syncthreads();

    // ---- GEMM3: M = tanh(x2 @ W3.T + b3)   (16x128)x(128x448), 28 N-tiles
    // Uniform 4-iteration unrolled loop: waves whose raw tile id >= 28 compute
    // a throwaway duplicate of tile 27 (EXEC stays all-1s) and skip the store.
    // 4 independent accumulation chains let the scheduler hide WMMA hazards.
#pragma unroll
    for (int it = 0; it < 4; ++it) {
      int ntr = wv + 8 * it;
      int nt  = (ntr < 28) ? ntr : 27;
      v8f acc = {};
#pragma unroll
      for (int kc = 0; kc < 4; ++kc)
        acc = wmma_bf16(&x2A[(kc * 32 + lane) * 16],
                        &W3B[((kc * 28 + nt) * 32 + lane) * 16], acc);
      if (ntr < 28) {   // wave-uniform branch
#pragma unroll
        for (int r = 0; r < 8; ++r) {
          int row = r + mbase, col = nt * 16 + lhalf;
          g3[row * 456 + col] = fast_tanh(acc[r] + b3L[col]);
        }
      }
    }
    __syncthreads();

    // ---- Euler update: h += act * dt * sum_c M[row][h][c] * dxdt[row][c]
#pragma unroll
    for (int i = 0; i < 4; ++i) {
      int p = tid * 4 + i;
      int row = p >> 6, hh = p & 63;
      const float* gr = &g3[row * 456 + hh * CC];
      const float* dx = &dxdtL[row * 8];
      float dh = 0.0f;
#pragma unroll
      for (int c = 0; c < CC; ++c) dh += gr[c] * dx[c];
      float hn = h32[row * 68 + hh] + actL[row] * dtL[row] * dh;
      h32[row * 68 + hh] = hn;
      hA[a_idx(row, hh)] = f32_to_bf16(hn);
    }
    __syncthreads();

    // ---- fused readout: pred = relu(h @ Wr1.T + br1) @ Wr2.T + br2
    if (wv < 2) {                      // full-wave EXEC inside
      v8f acc = {};
#pragma unroll
      for (int kc = 0; kc < 2; ++kc)
        acc = wmma_bf16(&hA[(kc * 32 + lane) * 16],
                        &WrB[((kc * 2 + wv) * 32 + lane) * 16], acc);
      int   j  = wv * 16 + lhalf;
      float w2 = wr2L[j], bb = br1L[j];
#pragma unroll
      for (int r = 0; r < 8; ++r) {
        float v = fmaxf(acc[r] + bb, 0.0f) * w2;
        v += __shfl_xor(v, 1);
        v += __shfl_xor(v, 2);
        v += __shfl_xor(v, 4);
        v += __shfl_xor(v, 8);         // sum over the 16 j's in this tile
        if (lhalf == 0) predP[wv * 16 + mbase + r] = v;
      }
    }
    __syncthreads();
    if (tid < TB)
      out[(size_t)(b0 + tid) * TT + t] = predP[tid] + predP[16 + tid] + br2L;
    __syncthreads();
  }
}

extern "C" void kernel_launch(void* const* d_in, const int* in_sizes, int n_in,
                              void* d_out, int out_size, void* d_ws, size_t ws_size,
                              hipStream_t stream) {
  (void)in_sizes; (void)n_in; (void)out_size; (void)d_ws; (void)ws_size;
  const float* timeg  = (const float*)d_in[0];
  const float* feat   = (const float*)d_in[1];
  // d_in[2] = mask (unused by forward)
  const int*   length = (const int*)  d_in[3];
  const float* Wp  = (const float*)d_in[4];
  const float* bp  = (const float*)d_in[5];
  const float* W1  = (const float*)d_in[6];
  const float* b1  = (const float*)d_in[7];
  const float* W2  = (const float*)d_in[8];
  const float* b2  = (const float*)d_in[9];
  const float* W3  = (const float*)d_in[10];
  const float* b3  = (const float*)d_in[11];
  const float* Wr1 = (const float*)d_in[12];
  const float* br1 = (const float*)d_in[13];
  const float* Wr2 = (const float*)d_in[14];
  const float* br2 = (const float*)d_in[15];
  float* out = (float*)d_out;

  dim3 grid(BATCH / TB);    // 256 workgroups, one 16-row batch tile each
  dim3 block(NTHREADS);     // 8 waves of 32
  ncde_scan_kernel<<<grid, block, 0, stream>>>(timeg, feat, length,
                                               Wp, bp, W1, b1, W2, b2, W3, b3,
                                               Wr1, br1, Wr2, br2, out);
}
